// SpikingNeuralNetwork_57286273795038
// MI455X (gfx1250) — matrix operand order repaired and checked by reading, not compiled, not run on hardware
//
#include <hip/hip_runtime.h>
#include <hip/hip_bf16.h>

// ---------------- types for WMMA (gfx1250, wave32) ----------------
typedef __bf16 bf16;
typedef __attribute__((ext_vector_type(16))) __bf16 v16bf;
typedef __attribute__((ext_vector_type(8)))  __bf16 v8bf;
typedef __attribute__((ext_vector_type(8)))  float  v8f;

#define ALPHA_C 0.818730753f /* exp(-0.2)  syn decay */
#define BETA_C  0.904837418f /* exp(-0.1)  mem decay */
#define REG_C   1e-7f

union AB16 { v16bf v; v8bf h[2]; };

// A fragment: 16x32 bf16, row-major source, row stride ldA (elements).
// Layout: lanes 0-15 -> M=lane, K=0..15 contiguous;
//         lanes 16-31 -> M=lane-16, K=16..31 (matches B fragment).
static __device__ inline v16bf load_frag_A(const bf16* A, int ldA, int tileM, int k0) {
    int lane = threadIdx.x & 31;
    int row  = lane & 15;
    int koff = (lane & 16) ? 16 : 0;
    const bf16* p = A + (size_t)(tileM * 16 + row) * (size_t)ldA + (size_t)(k0 + koff);
    AB16 u;
    u.h[0] = *(const v8bf*)(p);        // K = koff .. koff+7
    u.h[1] = *(const v8bf*)(p + 8);    // K = koff+8 .. koff+15
    return u.v;
}

// B fragment: 32x16 bf16 where B[k][n] = W[n][k], W row-major with stride ldW.
// Layout used here (same k order as A): lanes 0-15 -> N=lane, K=0..15 contiguous; lanes 16-31 -> K=16..31.
static __device__ inline v16bf load_frag_B(const bf16* W, int ldW, int tileN, int k0) {
    int lane = threadIdx.x & 31;
    int n    = lane & 15;
    int kb   = (lane & 16) ? 16 : 0;
    const bf16* p = W + (size_t)(tileN * 16 + n) * (size_t)ldW + (size_t)(k0 + kb);
    return *(const v16bf*)(p);
}

// ---------------- conversion / init kernels ----------------
__global__ void cvt_pad_kernel(const float* __restrict__ src, bf16* __restrict__ dst,
                               int R, int K, int Rp, int Kp) {
    int i = blockIdx.x * blockDim.x + threadIdx.x;
    if (i >= Rp * Kp) return;
    int r = i / Kp, k = i - r * Kp;
    float v = (r < R && k < K) ? src[(size_t)r * K + k] : 0.0f;
    dst[i] = (bf16)v;
}

// spike_trains (B,T,I) f32 -> spk_bf [t][b][k<Ip] bf16, zero-padded K
__global__ void cvt_spikes_kernel(const float* __restrict__ spikes, bf16* __restrict__ dst,
                                  int B, int T, int I, int Ip) {
    unsigned i = blockIdx.x * blockDim.x + threadIdx.x;
    unsigned n = (unsigned)T * B * Ip;
    if (i >= n) return;
    int k = i % Ip;
    unsigned tb = i / Ip;
    int b = tb % B, t = tb / B;
    float v = (k < I) ? spikes[((size_t)b * T + t) * (size_t)I + k] : 0.0f;
    dst[i] = (bf16)v;
}

__global__ void zero_f32_kernel(float* p, int n) {
    int i = blockIdx.x * blockDim.x + threadIdx.x;
    if (i < n) p[i] = 0.0f;
}

// ---------------- generic WMMA GEMM: C[M,N] = A[M,K] * W[N,K]^T ----------------
// Register-blocked over N: each wave computes a 16 x (16*NT) strip, reusing one
// A fragment for NT WMMA ops per K-step (cuts A traffic NT x).
template <int NT>
__global__ void gemm_bf16_wmma(const bf16* __restrict__ A, const bf16* __restrict__ W,
                               float* __restrict__ C, int M, int N, int K,
                               int ldA, int ldW) {
    int wave    = blockIdx.x * (blockDim.x >> 5) + (threadIdx.x >> 5);
    int ngroups = N / (16 * NT);
    int tileM   = wave / ngroups;
    int tileN0  = (wave - tileM * ngroups) * NT;
    if (tileM * 16 >= M) return;              // wave-uniform
    v8f c[NT];
#pragma unroll
    for (int j = 0; j < NT; ++j) c[j] = (v8f){};
    for (int k0 = 0; k0 < K; k0 += 32) {
        v16bf a = load_frag_A(A, ldA, tileM, k0);
        if (k0 + 32 < K)
            __builtin_prefetch(A + (size_t)(tileM * 16 + (threadIdx.x & 15)) * ldA + k0 + 32, 0, 1);
#pragma unroll
        for (int j = 0; j < NT; ++j) {
            v16bf b = load_frag_B(W, ldW, tileN0 + j, k0);
            c[j] = __builtin_amdgcn_wmma_f32_16x16x32_bf16(false, a, false, b,
                                                           (short)0, c[j], false, false);
        }
    }
    int lane = threadIdx.x & 31;
    int mofs = (lane & 16) ? 8 : 0;
#pragma unroll
    for (int j = 0; j < NT; ++j) {
        int n = (tileN0 + j) * 16 + (lane & 15);
#pragma unroll
        for (int r = 0; r < 8; ++r) {
            int m = tileM * 16 + mofs + r;
            C[(size_t)m * N + n] = c[j][r];
        }
    }
}

// ---------------- layer 0: elementwise LIF update ----------------
__global__ void layer0_step_kernel(const float* __restrict__ cur0_t,
                                   float* __restrict__ m0, float* __restrict__ s0,
                                   bf16* __restrict__ spk0_t, int n) {
    int i = blockIdx.x * blockDim.x + threadIdx.x;
    if (i >= n) return;
    float mm = m0[i], ss = s0[i];
    float spk = ((mm - 1.0f) > 0.0f) ? 1.0f : 0.0f;
    m0[i] = (BETA_C * mm + ss) * (1.0f - spk);
    s0[i] = ALPHA_C * ss + cur0_t[i];
    spk0_t[i] = (bf16)spk;
}

// ---------------- layer 1: dual WMMA GEMM + fused LIF epilogue ----------------
// cur1 = spk0_t @ W1^T + spk1_prev @ Wrec^T ; then recurrent LIF state update.
// NT=2 N-blocking: keeps 512 waves of parallelism on the serial critical path.
__global__ void layer1_step_kernel(const bf16* __restrict__ spk0_t,
                                   const bf16* __restrict__ spk1_prev, int has_prev,
                                   const bf16* __restrict__ W1, const bf16* __restrict__ Wrec,
                                   float* __restrict__ m1, float* __restrict__ s1,
                                   bf16* __restrict__ spk1_t, int H) {
    const int NT = 2;
    int wave    = blockIdx.x * (blockDim.x >> 5) + (threadIdx.x >> 5);
    int ngroups = H / (16 * NT);
    int tileM   = wave / ngroups;
    int tileN0  = (wave - tileM * ngroups) * NT;
    v8f c[NT];
#pragma unroll
    for (int j = 0; j < NT; ++j) c[j] = (v8f){};
    for (int k0 = 0; k0 < H; k0 += 32) {
        v16bf a = load_frag_A(spk0_t, H, tileM, k0);
#pragma unroll
        for (int j = 0; j < NT; ++j) {
            v16bf b = load_frag_B(W1, H, tileN0 + j, k0);
            c[j] = __builtin_amdgcn_wmma_f32_16x16x32_bf16(false, a, false, b,
                                                           (short)0, c[j], false, false);
        }
        if (has_prev) {  // wave-uniform scalar branch: EXEC stays all-ones
            v16bf a2 = load_frag_A(spk1_prev, H, tileM, k0);
#pragma unroll
            for (int j = 0; j < NT; ++j) {
                v16bf b2 = load_frag_B(Wrec, H, tileN0 + j, k0);
                c[j] = __builtin_amdgcn_wmma_f32_16x16x32_bf16(false, a2, false, b2,
                                                               (short)0, c[j], false, false);
            }
        }
    }
    int lane = threadIdx.x & 31;
    int mofs = (lane & 16) ? 8 : 0;
#pragma unroll
    for (int j = 0; j < NT; ++j) {
        int n = (tileN0 + j) * 16 + (lane & 15);
#pragma unroll
        for (int r = 0; r < 8; ++r) {
            int bidx = tileM * 16 + mofs + r;                 // batch index
            size_t idx = (size_t)bidx * H + n;                // (b, h)
            float mm = m1[idx], ss = s1[idx];
            float spk = ((mm - 1.0f) > 0.0f) ? 1.0f : 0.0f;   // spike from pre-update m1
            m1[idx] = (BETA_C * mm + ss) * (1.0f - spk);
            s1[idx] = ALPHA_C * ss + c[j][r];
            spk1_t[idx] = (bf16)spk;                          // recorded + sp1 for next step
        }
    }
}

// ---------------- readout double-exponential filter ----------------
__global__ void readout_filter_kernel(const float* __restrict__ h2p, const float* __restrict__ bout,
                                      float* __restrict__ out, int B, int T, int O, int ldh) {
    int i = blockIdx.x * blockDim.x + threadIdx.x;
    if (i >= B * O) return;
    int b = i / O, o = i - b * O;
    float bo = bout[o];
    float flt = 0.0f, ov = 0.0f;
    out[((size_t)b * (T + 1)) * O + o] = bo;              // t = 0 row is zeros (+bias)
    for (int t = 0; t < T; ++t) {
        float h2 = h2p[((size_t)t * B + b) * (size_t)ldh + o];
        float no = BETA_C * ov + flt;                     // uses OLD flt
        flt = ALPHA_C * flt + h2;
        ov = no;
        out[((size_t)b * (T + 1) + t + 1) * O + o] = ov + bo;
    }
}

// ---------------- spike regularization (deterministic 2-stage) ----------------
// Stage 1: block g sums rows [g*Jc, (g+1)*Jc) into partial[g][h]; fully coalesced
// (each warp reads 64B contiguous bf16 per row; thread owns columns tid, tid+256).
__global__ void neuron_partial_kernel(const bf16* __restrict__ spk_rec,
                                      float* __restrict__ partial, int TB, int H, int Jc) {
    int g  = blockIdx.x;
    int j0 = g * Jc;
    int j1 = (j0 + Jc < TB) ? (j0 + Jc) : TB;
    float acc0 = 0.0f, acc1 = 0.0f;
    for (int j = j0; j < j1; ++j) {
        const bf16* row = spk_rec + (size_t)j * H;
        acc0 += (float)row[threadIdx.x];
        acc1 += (float)row[threadIdx.x + 256];
    }
    partial[(size_t)g * H + threadIdx.x]       = acc0;
    partial[(size_t)g * H + threadIdx.x + 256] = acc1;
}

// Stage 2: pn[h] = sum_g partial[g][h] in fixed order (deterministic).
__global__ void neuron_reduce_kernel(const float* __restrict__ partial,
                                     float* __restrict__ pn, int G, int H) {
    int h = blockIdx.x * blockDim.x + threadIdx.x;
    if (h >= H) return;
    float a = 0.0f;
    for (int g = 0; g < G; ++g) a += partial[(size_t)g * H + h];
    pn[h] = a;
}

__global__ void finalize_reg_kernel(const float* __restrict__ pn, float* __restrict__ reg_out) {
    __shared__ float sS[512];
    __shared__ float sQ[512];
    int h = threadIdx.x;
    float p0 = pn[h], p1 = pn[512 + h];
    sS[h] = p0 + p1;
    sQ[h] = p0 * p0 + p1 * p1;
    __syncthreads();
    for (int s = 256; s > 0; s >>= 1) {
        if (h < s) { sS[h] += sS[h + s]; sQ[h] += sQ[h + s]; }
        __syncthreads();
    }
    if (h == 0) {
        const float inv_tbh = 1.0f / (100.0f * 512.0f * 512.0f);
        reg_out[0] = REG_C * (sS[0] * inv_tbh) + REG_C * (sQ[0] * (1.0f / 512.0f));
    }
}

// ---------------- host launch ----------------
extern "C" void kernel_launch(void* const* d_in, const int* in_sizes, int n_in,
                              void* d_out, int out_size, void* d_ws, size_t ws_size,
                              hipStream_t stream) {
    const int B = 512, T = 100, I = 700, Ip = 704, H = 512, O = 20, Op = 32;
    const int BH = B * H;          // 262144
    const int TB = T * B;          // 51200
    const int Jc = 512, G = TB / Jc;  // 100 partial blocks per layer

    const float* spikes = (const float*)d_in[0];   // (B,T,I)
    const float* W0     = (const float*)d_in[1];   // (H,I)
    const float* W1     = (const float*)d_in[2];   // (H,H)
    const float* Wrec   = (const float*)d_in[3];   // (H,H)
    const float* Wout   = (const float*)d_in[4];   // (O,H)
    const float* bout   = (const float*)d_in[5];   // (O,)
    float* out = (float*)d_out;                    // (B,T+1,O) then reg scalar

    // ---- carve 256B-aligned workspace ----
    size_t off = 0;
    auto carve = [&](size_t bytes) -> char* {
        char* p = (char*)d_ws + off;
        off += (bytes + 255) & ~(size_t)255;
        return p;
    };
    bf16*  spk_bf   = (bf16*)carve((size_t)TB * Ip * 2);    // [t][b][i] padded
    bf16*  w0_bf    = (bf16*)carve((size_t)H * Ip * 2);
    bf16*  w1_bf    = (bf16*)carve((size_t)H * H * 2);
    bf16*  wrec_bf  = (bf16*)carve((size_t)H * H * 2);
    bf16*  wout_bf  = (bf16*)carve((size_t)Op * H * 2);     // rows 20..31 zero
    float* cur0     = (float*)carve((size_t)TB * H * 4);    // [t][b][h]
    float* m0       = (float*)carve((size_t)BH * 4);
    float* s0       = (float*)carve((size_t)BH * 4);
    float* m1       = (float*)carve((size_t)BH * 4);
    float* s1       = (float*)carve((size_t)BH * 4);
    bf16*  spk0_rec = (bf16*)carve((size_t)TB * H * 2);     // [t][b][h]
    bf16*  spk1_rec = (bf16*)carve((size_t)TB * H * 2);
    float* h2p      = (float*)carve((size_t)TB * Op * 4);   // [t*b][32]
    float* part0    = (float*)carve((size_t)G * H * 4);
    float* part1    = (float*)carve((size_t)G * H * 4);
    float* pn       = (float*)carve(1024 * 4);              // per-neuron sums, 2 layers
    (void)ws_size; (void)in_sizes; (void)n_in;

    // ---- conversions ----
    {
        unsigned n = (unsigned)TB * Ip;
        cvt_spikes_kernel<<<(n + 255) / 256, 256, 0, stream>>>(spikes, spk_bf, B, T, I, Ip);
    }
    cvt_pad_kernel<<<(H * Ip + 255) / 256, 256, 0, stream>>>(W0, w0_bf, H, I, H, Ip);
    cvt_pad_kernel<<<(H * H + 255) / 256, 256, 0, stream>>>(W1, w1_bf, H, H, H, H);
    cvt_pad_kernel<<<(H * H + 255) / 256, 256, 0, stream>>>(Wrec, wrec_bf, H, H, H, H);
    cvt_pad_kernel<<<(Op * H + 255) / 256, 256, 0, stream>>>(Wout, wout_bf, O, H, Op, H);

    // ---- zero LIF state ----
    zero_f32_kernel<<<(BH + 255) / 256, 256, 0, stream>>>(m0, BH);
    zero_f32_kernel<<<(BH + 255) / 256, 256, 0, stream>>>(s0, BH);
    zero_f32_kernel<<<(BH + 255) / 256, 256, 0, stream>>>(m1, BH);
    zero_f32_kernel<<<(BH + 255) / 256, 256, 0, stream>>>(s1, BH);

    // ---- big time-parallel GEMM: cur0[t*B+b, h] = spikes @ W0^T (NT=4) ----
    {
        int waves = (TB / 16) * (H / 64);          // 3200 * 8 = 25600
        gemm_bf16_wmma<4><<<waves / 8, 256, 0, stream>>>(spk_bf, w0_bf, cur0, TB, H, Ip, Ip, Ip);
    }

    // ---- sequential recurrent loop ----
    {
        int waves1 = (B / 16) * (H / 32);          // 32 * 16 = 512 -> 64 blocks
        for (int t = 0; t < T; ++t) {
            layer0_step_kernel<<<(BH + 255) / 256, 256, 0, stream>>>(
                cur0 + (size_t)t * BH, m0, s0, spk0_rec + (size_t)t * BH, BH);
            const bf16* prev = (t > 0) ? (spk1_rec + (size_t)(t - 1) * BH) : spk1_rec;
            layer1_step_kernel<<<waves1 / 8, 256, 0, stream>>>(
                spk0_rec + (size_t)t * BH, prev, (t > 0) ? 1 : 0,
                w1_bf, wrec_bf, m1, s1, spk1_rec + (size_t)t * BH, H);
        }
    }

    // ---- readout projection (WMMA, O padded to 32, NT=2) + filter ----
    {
        int waves = (TB / 16) * (Op / 32);         // 3200 -> 400 blocks
        gemm_bf16_wmma<2><<<waves / 8, 256, 0, stream>>>(spk1_rec, wout_bf, h2p, TB, Op, H, H, H);
    }
    readout_filter_kernel<<<(B * O + 255) / 256, 256, 0, stream>>>(h2p, bout, out, B, T, O, Op);

    // ---- regularization loss (deterministic 2-stage, coalesced) ----
    neuron_partial_kernel<<<G, 256, 0, stream>>>(spk0_rec, part0, TB, H, Jc);
    neuron_partial_kernel<<<G, 256, 0, stream>>>(spk1_rec, part1, TB, H, Jc);
    neuron_reduce_kernel<<<(H + 255) / 256, 256, 0, stream>>>(part0, pn, G, H);
    neuron_reduce_kernel<<<(H + 255) / 256, 256, 0, stream>>>(part1, pn + 512, G, H);
    finalize_reg_kernel<<<1, 512, 0, stream>>>(pn, out + (size_t)B * (T + 1) * O);
}
